// RNN_4372276707338
// MI455X (gfx1250) — matrix operand order repaired and checked by reading, not compiled, not run on hardware
//
#include <hip/hip_runtime.h>
#include <hip/hip_bf16.h>

#define SEQ_LEN 2048
#define BATCH   64
#define INPUT   256
#define HIDDEN  256
#define OUTPUT  256

typedef __bf16 bf16;
typedef __attribute__((ext_vector_type(16))) __bf16 v16bf;
typedef __attribute__((ext_vector_type(8)))  float  v8f;
typedef int v4i __attribute__((ext_vector_type(4)));
typedef unsigned int u32x4 __attribute__((ext_vector_type(4)));
typedef int i32x4 __attribute__((ext_vector_type(4)));
typedef int i32x8 __attribute__((ext_vector_type(8)));

// Padded LDS row stride (elements). 264*2B = 528B = 132 dwords == 4 (mod 64):
// consecutive rows land on rotated banks -> conflict-free b128 fragment reads.
#define LDA 264

// ---- CDNA5 async global<->LDS path (ASYNCcnt), guarded so compile never breaks
#if defined(__gfx1250__) && \
    __has_builtin(__builtin_amdgcn_global_load_async_to_lds_b128) && \
    __has_builtin(__builtin_amdgcn_global_store_async_from_lds_b128) && \
    __has_builtin(__builtin_amdgcn_s_wait_asynccnt)
#define RNN_HAVE_ASYNC 1
#else
#define RNN_HAVE_ASYNC 0
#endif

// ---- CDNA5 Tensor Data Mover path (TENSORcnt)
#if defined(__gfx1250__) && \
    __has_builtin(__builtin_amdgcn_tensor_load_to_lds) && \
    __has_builtin(__builtin_amdgcn_s_wait_tensorcnt)
#define RNN_HAVE_TDM 1
#else
#define RNN_HAVE_TDM 0
#endif

__device__ __forceinline__ void rnn_wait_async() {
#if RNN_HAVE_ASYNC
    __builtin_amdgcn_s_wait_asynccnt(0);
#endif
}

// Copy a 64x256 bf16 tile (row-major, stride HIDDEN) global -> LDS (stride LDA).
__device__ __forceinline__ void stage_tile_g2l(const bf16* __restrict__ src,
                                               bf16* dst, int tid, int nthreads) {
    for (int idx = tid; idx < 64 * 32; idx += nthreads) {
        const int r = idx >> 5, c8 = (idx & 31) * 8;
#if RNN_HAVE_ASYNC
        __builtin_amdgcn_global_load_async_to_lds_b128(
            (__attribute__((address_space(1))) v4i*)(src + r * HIDDEN + c8),
            (__attribute__((address_space(3))) v4i*)(dst + r * LDA + c8),
            0, 0);
#else
        *(uint4*)(dst + r * LDA + c8) = *(const uint4*)(src + r * HIDDEN + c8);
#endif
    }
}

// Copy a 64x256 bf16 tile LDS (stride LDA) -> global (stride HIDDEN).
__device__ __forceinline__ void stage_tile_l2g(bf16* __restrict__ dst,
                                               const bf16* src, int tid, int nthreads) {
    for (int idx = tid; idx < 64 * 32; idx += nthreads) {
        const int r = idx >> 5, c8 = (idx & 31) * 8;
#if RNN_HAVE_ASYNC
        __builtin_amdgcn_global_store_async_from_lds_b128(
            (__attribute__((address_space(1))) v4i*)(dst + r * HIDDEN + c8),
            (__attribute__((address_space(3))) v4i*)(src + r * LDA + c8),
            0, 0);
#else
        *(uint4*)(dst + r * HIDDEN + c8) = *(const uint4*)(src + r * LDA + c8);
#endif
    }
}

#if RNN_HAVE_TDM
// One TDM descriptor (ISA 8.3/8.4): DMA a 64-row x 256-col bf16 tile
// (global row stride 256 elems) into LDS, inserting 16B of pad every 512B
// so the LDS tile has row stride LDA=264 elems (bank-conflict-free layout).
// Issued by a single wave; completion via TENSORcnt.
__device__ __forceinline__ void tdm_load_tile(const bf16* src, bf16* dstLds) {
    const unsigned long long ga =
        (unsigned long long)(uintptr_t)src;
    const unsigned lds =
        (unsigned)(uintptr_t)(__attribute__((address_space(3))) bf16*)dstLds;

    u32x4 g0;
    g0[0] = 1u;                                   // count=1 (valid user D#)
    g0[1] = lds;                                  // lds_addr (bytes)
    g0[2] = (unsigned)(ga & 0xFFFFFFFFu);         // global_addr[31:0]
    g0[3] = (unsigned)((ga >> 32) & 0x01FFFFFFu)  // global_addr[56:32]
          | (2u << 30);                           // type = 2 ("image")

    i32x8 g1;
    g1[0] = (1 << 16)      // data_size: 2 bytes
          | (1 << 20)      // pad_enable
          | (6 << 22)      // pad_interval: 128 dwords (512B) between pads
          | (3 << 25);     // pad_amount: 4 dwords (16B)
    g1[1] = (256 << 16);   // tensor_dim0 = 256 (low16 in [31:16])
    g1[2] = (64 << 16);    // dim0_hi=0 | tensor_dim1 = 64
    g1[3] = (256 << 16);   // dim1_hi=0 | tile_dim0 = 256
    g1[4] = 64;            // tile_dim1 = 64 | tile_dim2 = 0
    g1[5] = 256;           // tensor_dim0_stride[31:0] = 256 (elements)
    g1[6] = 0;             // stride0_hi | tensor_dim1_stride lo (unused, 2-D)
    g1[7] = 0;

    const i32x4 z4 = {0, 0, 0, 0};
#if __clang_major__ >= 23
    const i32x8 z8 = {0, 0, 0, 0, 0, 0, 0, 0};
    __builtin_amdgcn_tensor_load_to_lds(g0, g1, z4, z4, z8, 0);
#else
    __builtin_amdgcn_tensor_load_to_lds(g0, g1, z4, z4, 0);
#endif
}
#endif

// ---------------------------------------------------------------------------
// Fragment loaders (layouts per CDNA5 ISA 7.12.2, wave32)
// A 16x32 bf16 from LDS: lane m=lane&15 holds row m; kh=(lane>>4)*8:
//   elems 0..7 -> K=kh+0..7 ; elems 8..15 -> K=16+kh+0..7   (two b128 loads)
// ---------------------------------------------------------------------------
__device__ __forceinline__ v16bf load_a_frag_lds(const bf16* buf, int m0, int k0, int lane) {
    const int m  = m0 + (lane & 15);
    const int kh = (lane >> 4) << 3;
    const bf16* p = buf + m * LDA + k0 + kh;
    v16bf f;
    ((uint4*)&f)[0] = *(const uint4*)(p);
    ((uint4*)&f)[1] = *(const uint4*)(p + 16);
    return f;
}

// B 32x16 bf16 from TRANSPOSED weights (N x K row-major):
// lane col n=n0+(lane&15); kh=(lane>>4)*8: elems 0..7 -> K=k0+kh+0..7 ; elems 8..15 -> K=k0+16+kh+0..7.
// Two global_load_b128 per lane.
__device__ __forceinline__ v16bf load_b_frag_t(const bf16* __restrict__ BT,
                                               int k0, int n0, int lane) {
    const int n  = n0 + (lane & 15);
    const int kb = k0 + ((lane >> 4) << 3);
    const bf16* p = BT + n * HIDDEN + kb;
    v16bf f;
    ((uint4*)&f)[0] = *(const uint4*)(p);
    ((uint4*)&f)[1] = *(const uint4*)(p + 16);
    return f;
}

// ---------------------------------------------------------------------------
// Kernel 0: convert + transpose weights f32 -> bf16 (stored N x K row-major)
// ---------------------------------------------------------------------------
__global__ void rnn_convert_weights(const float* __restrict__ Ux,
                                    const float* __restrict__ Wh,
                                    const float* __restrict__ Vo,
                                    bf16* __restrict__ Uxt,
                                    bf16* __restrict__ Wht,
                                    bf16* __restrict__ Vot) {
    int i = blockIdx.x * blockDim.x + threadIdx.x;
    if (i < HIDDEN * HIDDEN) {
        const int k = i >> 8, n = i & 255;      // read row-major (K x N), coalesced
        const int j = n * HIDDEN + k;           // write transposed
        Uxt[j] = (bf16)Ux[i];
        Wht[j] = (bf16)Wh[i];
        Vot[j] = (bf16)Vo[i];
    }
}

// ---------------------------------------------------------------------------
// Kernel 1: xp = X @ Ux + bh (stored bf16). 256 thr = 8 waves; 64 rows/block.
// ---------------------------------------------------------------------------
__global__ void __launch_bounds__(256)
rnn_xproj_kernel(const float* __restrict__ x, const bf16* __restrict__ Uxt,
                 const float* __restrict__ bh, bf16* __restrict__ xp) {
    __shared__ bf16 atile[64 * LDA];
    const int tid  = threadIdx.x;
    const int lane = tid & 31;
    const int wave = tid >> 5;
    const long row0 = (long)blockIdx.x * 64;

    // Stage + convert 64x256 f32 input tile -> bf16 LDS
    for (int idx = tid; idx < 64 * 64; idx += 256) {
        const int r  = idx >> 6;
        const int c4 = (idx & 63) * 4;
        const float4 v = *(const float4*)(x + (row0 + r) * INPUT + c4);
        bf16* d = &atile[r * LDA + c4];
        d[0] = (bf16)v.x; d[1] = (bf16)v.y; d[2] = (bf16)v.z; d[3] = (bf16)v.w;
    }
    __syncthreads();

    for (int nt = 0; nt < 2; ++nt) {
        const int n0 = (wave * 2 + nt) * 16;
        v16bf bfr[8];
#pragma unroll
        for (int k = 0; k < 8; ++k) bfr[k] = load_b_frag_t(Uxt, k * 32, n0, lane);
        const float bias = bh[n0 + (lane & 15)];
        const int colg = n0 + (lane & 15);
        for (int mt = 0; mt < 4; ++mt) {
            v8f acc;
#pragma unroll
            for (int r = 0; r < 8; ++r) acc[r] = bias;
#pragma unroll
            for (int k = 0; k < 8; ++k) {
                v16bf a = load_a_frag_lds(atile, mt * 16, k * 32, lane);
                acc = __builtin_amdgcn_wmma_f32_16x16x32_bf16(
                    false, a, false, bfr[k], (short)0, acc, false, false);
            }
            const int rbase = mt * 16 + ((lane >> 4) << 3);
#pragma unroll
            for (int r = 0; r < 8; ++r)
                xp[(row0 + rbase + r) * HIDDEN + colg] = (bf16)acc[r];
        }
    }
}

// ---------------------------------------------------------------------------
// Kernel 2: serial scan  h_t = tanh(xp_t + h_{t-1} @ Wh)
// One workgroup, 512 thr = 16 waves; wave w owns N-tile w*16.
// Wh fragments resident in VGPRs; h in LDS; xp_t double-buffered in LDS with
// TDM (or async) prefetch of t+1; h history streamed out async-from-LDS.
// ---------------------------------------------------------------------------
__global__ void __launch_bounds__(512)
rnn_scan_kernel(const float* __restrict__ h0, const bf16* __restrict__ Wht,
                const bf16* __restrict__ xp, bf16* __restrict__ hhist,
                float* __restrict__ hfinal) {
    __shared__ bf16 hbuf[64 * LDA];
    __shared__ bf16 xbuf[2][64 * LDA];

    const int tid  = threadIdx.x;
    const int lane = tid & 31;
    const int wave = tid >> 5;          // 0..15
    const int n0   = wave * 16;
    const int col  = lane & 15;
    const int roff = (lane >> 4) << 3;

    // Persistent Wh fragments: 8 K-steps x 8 VGPRs = 64 VGPRs/lane
    v16bf wfr[8];
#pragma unroll
    for (int k = 0; k < 8; ++k) wfr[k] = load_b_frag_t(Wht, k * 32, n0, lane);

    // Prologue: stage xp[0] and init h from h0 (f32 -> bf16)
#if RNN_HAVE_TDM
    if (tid < 32) tdm_load_tile(xp, &xbuf[0][0]);
#else
    stage_tile_g2l(xp, &xbuf[0][0], tid, 512);
#endif
    for (int idx = tid; idx < 64 * 256; idx += 512) {
        const int r = idx >> 8, c = idx & 255;
        hbuf[r * LDA + c] = (bf16)h0[r * 256 + c];
    }
#if RNN_HAVE_TDM
    if (tid < 32) __builtin_amdgcn_s_wait_tensorcnt(0);
#endif
    rnn_wait_async();
    __syncthreads();

    int xb = 0;
    for (int t = 0; t < SEQ_LEN; ++t) {
        // Kick off prefetch of xp[t+1] into the other buffer (TDM: wave 0 only)
        if (t + 1 < SEQ_LEN) {
#if RNN_HAVE_TDM
            if (tid < 32)
                tdm_load_tile(xp + (size_t)(t + 1) * (BATCH * HIDDEN),
                              &xbuf[xb ^ 1][0]);
#else
            stage_tile_g2l(xp + (size_t)(t + 1) * (BATCH * HIDDEN),
                           &xbuf[xb ^ 1][0], tid, 512);
#endif
        }

        // Compute phase: acc = xp_t + h_{t-1} @ Wh  (all 4 M-tiles in registers)
        const bf16* xcur = &xbuf[xb][0];
        v8f acc[4];
#pragma unroll
        for (int mt = 0; mt < 4; ++mt) {
            const int rbase = mt * 16 + roff;
#pragma unroll
            for (int r = 0; r < 8; ++r)
                acc[mt][r] = (float)xcur[(rbase + r) * LDA + n0 + col];
#pragma unroll
            for (int k = 0; k < 8; ++k) {
                v16bf a = load_a_frag_lds(hbuf, mt * 16, k * 32, lane);
                acc[mt] = __builtin_amdgcn_wmma_f32_16x16x32_bf16(
                    false, a, false, wfr[k], (short)0, acc[mt], false, false);
            }
        }
        __syncthreads();   // all reads of h_{t-1} complete
        // Drain async h-store(t-1): hbuf safe to overwrite.
        rnn_wait_async();
#if RNN_HAVE_TDM
        // Wave 0 confirms xp[t+1] landed; barrier below publishes to all waves.
        if (tid < 32) __builtin_amdgcn_s_wait_tensorcnt(0);
#endif

        // Write phase: tanh -> LDS h (+ final state at t = S-1)
#pragma unroll
        for (int mt = 0; mt < 4; ++mt) {
            const int rbase = mt * 16 + roff;
#pragma unroll
            for (int r = 0; r < 8; ++r) {
                const float hv = tanhf(acc[mt][r]);
                hbuf[(rbase + r) * LDA + n0 + col] = (bf16)hv;
                if (t == SEQ_LEN - 1)
                    hfinal[(rbase + r) * HIDDEN + n0 + col] = hv;
            }
        }
        __syncthreads();   // h_t complete in LDS; xbuf[xb^1] valid for all

        // Stream h_t to HBM asynchronously; overlaps next step's compute.
        stage_tile_l2g(hhist + (size_t)t * (BATCH * HIDDEN), hbuf, tid, 512);
        xb ^= 1;
    }
    rnn_wait_async();
}

// ---------------------------------------------------------------------------
// Kernel 3: out = h_hist @ Vo + co   (f32 output)
// ---------------------------------------------------------------------------
__global__ void __launch_bounds__(256)
rnn_outproj_kernel(const bf16* __restrict__ h, const bf16* __restrict__ Vot,
                   const float* __restrict__ co, float* __restrict__ out) {
    __shared__ bf16 atile[64 * LDA];
    const int tid  = threadIdx.x;
    const int lane = tid & 31;
    const int wave = tid >> 5;
    const long row0 = (long)blockIdx.x * 64;

    stage_tile_g2l(h + row0 * HIDDEN, atile, tid, 256);
    rnn_wait_async();
    __syncthreads();

    for (int nt = 0; nt < 2; ++nt) {
        const int n0 = (wave * 2 + nt) * 16;
        v16bf bfr[8];
#pragma unroll
        for (int k = 0; k < 8; ++k) bfr[k] = load_b_frag_t(Vot, k * 32, n0, lane);
        const float bias = co[n0 + (lane & 15)];
        const int colg = n0 + (lane & 15);
        for (int mt = 0; mt < 4; ++mt) {
            v8f acc;
#pragma unroll
            for (int r = 0; r < 8; ++r) acc[r] = bias;
#pragma unroll
            for (int k = 0; k < 8; ++k) {
                v16bf a = load_a_frag_lds(atile, mt * 16, k * 32, lane);
                acc = __builtin_amdgcn_wmma_f32_16x16x32_bf16(
                    false, a, false, bfr[k], (short)0, acc, false, false);
            }
            const int rbase = mt * 16 + ((lane >> 4) << 3);
#pragma unroll
            for (int r = 0; r < 8; ++r)
                out[(row0 + rbase + r) * OUTPUT + colg] = acc[r];
        }
    }
}

// ---------------------------------------------------------------------------
extern "C" void kernel_launch(void* const* d_in, const int* in_sizes, int n_in,
                              void* d_out, int out_size, void* d_ws, size_t ws_size,
                              hipStream_t stream) {
    (void)in_sizes; (void)n_in; (void)out_size; (void)ws_size;

    const float* x  = (const float*)d_in[0];   // (S,B,I)
    const float* h0 = (const float*)d_in[1];   // (B,H)
    const float* Wh = (const float*)d_in[2];   // (H,H)
    const float* Ux = (const float*)d_in[3];   // (I,H)
    const float* bh = (const float*)d_in[4];   // (H,)
    const float* Vo = (const float*)d_in[5];   // (H,O)
    const float* co = (const float*)d_in[6];   // (O,)

    float* out    = (float*)d_out;                              // (S,B,O)
    float* hfinal = out + (size_t)SEQ_LEN * BATCH * OUTPUT;     // (B,H)

    // Workspace layout (transposed bf16 weights + activations)
    char* ws = (char*)d_ws;
    bf16* Uxt = (bf16*)ws;  ws += (size_t)HIDDEN * HIDDEN * sizeof(bf16);
    bf16* Wht = (bf16*)ws;  ws += (size_t)HIDDEN * HIDDEN * sizeof(bf16);
    bf16* Vot = (bf16*)ws;  ws += (size_t)HIDDEN * HIDDEN * sizeof(bf16);
    bf16* xpb = (bf16*)ws;  ws += (size_t)SEQ_LEN * BATCH * HIDDEN * sizeof(bf16);
    bf16* hh  = (bf16*)ws;  // (S,B,H) bf16 history

    rnn_convert_weights<<<(HIDDEN * HIDDEN + 255) / 256, 256, 0, stream>>>(
        Ux, Wh, Vo, Uxt, Wht, Vot);

    rnn_xproj_kernel<<<(SEQ_LEN * BATCH) / 64, 256, 0, stream>>>(x, Uxt, bh, xpb);

    rnn_scan_kernel<<<1, 512, 0, stream>>>(h0, Wht, xpb, hh, hfinal);

    rnn_outproj_kernel<<<(SEQ_LEN * BATCH) / 64, 256, 0, stream>>>(hh, Vot, co, out);
}
